// Decoder_38293928411158
// MI455X (gfx1250) — matrix operand order repaired and checked by reading, not compiled, not run on hardware
//
#include <hip/hip_runtime.h>
#include <hip/hip_bf16.h>

typedef unsigned short u16;
typedef unsigned int   u32;

typedef __attribute__((ext_vector_type(16))) __bf16 v16bf;
typedef __attribute__((ext_vector_type(8)))  float  v8f;

#define NBLK 6
#define HH   12
#define DD   768
#define EE   32
#define LL   64
#define SS   512
#define BB   64
#define DFF  3072           // 4*D
#define M1   (BB*LL)        // 4096
#define M2   (BB*SS)        // 32768

__device__ __forceinline__ u16 f2bf(float f) {
  u32 x = __float_as_uint(f);
  u32 r = x + 0x7fffu + ((x >> 16) & 1u);
  return (u16)(r >> 16);
}

__device__ __forceinline__ float gelu_exact(float x) {
  return 0.5f * x * (1.0f + erff(x * 0.70710678118654752f));
}

union Frag { v16bf v; uint4 q[2]; u16 u[16]; };
union Acc  { v8f   v; float f[8]; };

// A-matrix 16x32 bf16 fragment: lane (m = lane&15, g = lane>>4) holds
// K = k0+8g+(0..7) in elems 0..7 and K = k0+16+8g+(0..7) in elems 8..15.
__device__ __forceinline__ void ldfragA(Frag& f, const u16* p) {
  f.q[0] = *reinterpret_cast<const uint4*>(p);        // +0  elements
  f.q[1] = *reinterpret_cast<const uint4*>(p + 16);   // +16 elements
}
// B-matrix 32x16 bf16 fragment (from BT row-major): lane (n, g) holds
// K = k0+8g+(0..7) in elems 0..7 and K = k0+16+8g+(0..7) in elems 8..15.
__device__ __forceinline__ void ldfragB(Frag& f, const u16* p) {
  f.q[0] = *reinterpret_cast<const uint4*>(p);
  f.q[1] = *reinterpret_cast<const uint4*>(p + 16);
}

__device__ __forceinline__ v8f wmma_bf16(const Frag& a, const Frag& b, v8f c) {
  return __builtin_amdgcn_wmma_f32_16x16x32_bf16(
      false, a.v, false, b.v, (short)0, c, false, false);
}

// ---------------------------------------------------------------------------
// elementwise fp32 -> bf16
__global__ __launch_bounds__(256) void cvt_bf16_k(const float* __restrict__ src,
                                                  u16* __restrict__ dst, long n) {
  long i = (long)blockIdx.x * 256 + threadIdx.x;
  if (i < n) dst[i] = f2bf(src[i]);
}

// transpose + convert: src [Z][Kk][Nn] fp32 -> dst [Z][Nn][Kk] bf16
__global__ __launch_bounds__(256) void tcvt_bf16_k(const float* __restrict__ src,
                                                   u16* __restrict__ dst,
                                                   int Kk, int Nn, long total) {
  long o = (long)blockIdx.x * 256 + threadIdx.x;
  if (o >= total) return;
  long mat = (long)Kk * Nn;
  long z = o / mat;
  long rem = o - z * mat;
  int n = (int)(rem / Kk);
  int k = (int)(rem - (long)n * Kk);
  dst[o] = f2bf(src[z * mat + (long)k * Nn + n]);
}

// x[b*L+q][d] = pos[q][d]  (f32 and bf16)
__global__ __launch_bounds__(256) void posinit_k(const float* __restrict__ pos,
                                                 float* __restrict__ xf,
                                                 u16* __restrict__ xb) {
  long i = (long)blockIdx.x * 256 + threadIdx.x;   // over 4096*768
  if (i >= (long)M1 * DD) return;
  int d = (int)(i % DD);
  int q = (int)((i / DD) % LL);
  float v = pos[(long)q * DD + d];
  xf[i] = v;
  xb[i] = f2bf(v);
}

// ---------------------------------------------------------------------------
// Generic bf16 WMMA GEMM: C[M,N] = A[M,Kd] * BT[N,Kd]^T
// wave tile 64 x (TN*16) (4 M-subtiles x TN N-subtiles), block = 4 waves.
// SMODE: 0=f32 out, 1=bf16 out, 2=bf16 out with per-b transpose (V^T layout)
// EPI:   0=none, 1=+Res[row*N+col] (f32), 2=gelu
template<int SMODE, int EPI, int TN>
__global__ __launch_bounds__(128)
void gemm_k(const u16* __restrict__ A, const u16* __restrict__ BT,
            void* __restrict__ Cout, const float* __restrict__ Res,
            int M, int N, int Kd, long aZ, long bZ, long cZ) {
  const int lane = threadIdx.x & 31;
  const int wave = threadIdx.x >> 5;
  const int lm = lane & 15;
  const int g  = lane >> 4;
  const u16* Ab = A  + (long)blockIdx.z * aZ;
  const u16* Bb = BT + (long)blockIdx.z * bZ;
  const int rowBase = blockIdx.x * 256 + wave * 64;
  const int colBase = blockIdx.y * (TN * 16);

  Acc acc[4][TN];
#pragma unroll
  for (int mi = 0; mi < 4; ++mi)
#pragma unroll
    for (int ni = 0; ni < TN; ++ni)
#pragma unroll
      for (int r = 0; r < 8; ++r) acc[mi][ni].f[r] = 0.f;

  for (int k0 = 0; k0 < Kd; k0 += 32) {
    Frag a[4], b[TN];
#pragma unroll
    for (int mi = 0; mi < 4; ++mi)
      ldfragA(a[mi], Ab + (long)(rowBase + mi * 16 + lm) * Kd + k0 + 8 * g);
#pragma unroll
    for (int ni = 0; ni < TN; ++ni)
      ldfragB(b[ni], Bb + (long)(colBase + ni * 16 + lm) * Kd + k0 + 8 * g);
#pragma unroll
    for (int mi = 0; mi < 4; ++mi)
#pragma unroll
      for (int ni = 0; ni < TN; ++ni)
        acc[mi][ni].v = wmma_bf16(a[mi], b[ni], acc[mi][ni].v);
  }

#pragma unroll
  for (int mi = 0; mi < 4; ++mi) {
#pragma unroll
    for (int ni = 0; ni < TN; ++ni) {
#pragma unroll
      for (int r = 0; r < 8; ++r) {
        int row = rowBase + mi * 16 + r + 8 * g;
        int col = colBase + ni * 16 + lm;
        float vv = acc[mi][ni].f[r];
        if (EPI == 1) vv += Res[(long)row * N + col];
        if (EPI == 2) vv = gelu_exact(vv);
        if (SMODE == 0) {
          ((float*)Cout)[(long)blockIdx.z * cZ + (long)row * N + col] = vv;
        } else if (SMODE == 1) {
          ((u16*)Cout)[(long)blockIdx.z * cZ + (long)row * N + col] = f2bf(vv);
        } else {
          int bb = row >> 9;            // row = b*512 + s
          int ss = row & 511;
          ((u16*)Cout)[(long)blockIdx.z * cZ + (long)bb * (EE * SS)
                       + (long)col * SS + ss] = f2bf(vv);
        }
      }
    }
  }
}

// ---------------------------------------------------------------------------
// Fused attention per (b,h): scores = Q K^T * scale -> softmax -> O = P V
// Q: [H][4096][32] bf16, K: [H][32768][32] bf16, Vt: [H][B][32][512] bf16
// Oc: [4096][384] bf16 (row = b*64+q, col = h*32+e)
__global__ __launch_bounds__(256)
void attn_k(const u16* __restrict__ Q, const u16* __restrict__ Kx,
            const u16* __restrict__ Vt, u16* __restrict__ Oc) {
  const int bh = blockIdx.x;
  const int b = bh / HH, h = bh % HH;
  extern __shared__ char smem[];
  float* sc      = (float*)smem;                              // [64][512] f32
  u16*   at      = (u16*)(smem + LL * SS * 4);                // [64][512] bf16
  float* red     = (float*)(smem + LL * SS * 4 + LL * SS * 2);// [256]
  float* rowstat = red + 256;                                 // [64]

  const int tid  = threadIdx.x;
  const int lane = tid & 31, wave = tid >> 5;
  const int lm = lane & 15, g = lane >> 4;

  // ---- phase 1: scores (M=64, N=512, K=32), 8 waves x 16 tiles
  {
    const u16* Qb = Q  + ((long)h * M1 + (long)b * LL) * EE;
    const u16* Kb = Kx + ((long)h * M2 + (long)b * SS) * EE;
    const int rs = (wave & 3) * 16;
    const int cb = (wave >> 2) * 256;
    Frag a; ldfragA(a, Qb + (rs + lm) * EE + 8 * g);
    const float scale = 0.03608439182435161f;  // 768^-0.5
    for (int ci = 0; ci < 16; ++ci) {
      int col = cb + ci * 16;
      Frag bf; ldfragB(bf, Kb + (col + lm) * EE + 8 * g);
      Acc acc;
#pragma unroll
      for (int r = 0; r < 8; ++r) acc.f[r] = 0.f;
      acc.v = wmma_bf16(a, bf, acc.v);
#pragma unroll
      for (int r = 0; r < 8; ++r)
        sc[(rs + r + 8 * g) * SS + col + lm] = acc.f[r] * scale;
    }
  }
  __syncthreads();

  // ---- phase 2: row softmax (4 threads per row)
  {
    int r = tid >> 2;
    int part = tid & 3;
    float* rowp = sc + r * SS;
    int j0 = part * 128;
    float mx = -3.4e38f;
    for (int j = j0; j < j0 + 128; ++j) mx = fmaxf(mx, rowp[j]);
    red[tid] = mx;
    __syncthreads();
    if (part == 0)
      rowstat[r] = fmaxf(fmaxf(red[tid], red[tid + 1]),
                         fmaxf(red[tid + 2], red[tid + 3]));
    __syncthreads();
    float m = rowstat[r];
    float s = 0.f;
    for (int j = j0; j < j0 + 128; ++j) {
      float e = expf(rowp[j] - m);
      rowp[j] = e;
      s += e;
    }
    red[tid] = s;
    __syncthreads();
    if (part == 0)
      rowstat[r] = red[tid] + red[tid + 1] + red[tid + 2] + red[tid + 3];
    __syncthreads();
    float inv = 1.0f / rowstat[r];
    for (int j = j0; j < j0 + 128; ++j)
      at[r * SS + j] = f2bf(rowp[j] * inv);
  }
  __syncthreads();

  // ---- phase 3: O = P V (M=64, N=32, K=512), 8 waves -> 8 tiles
  {
    const u16* Vb = Vt + ((long)h * BB + b) * (EE * SS);
    int mi = wave & 3, ni = wave >> 2;
    int rs = mi * 16;
    Acc acc;
#pragma unroll
    for (int r = 0; r < 8; ++r) acc.f[r] = 0.f;
    for (int k0 = 0; k0 < SS; k0 += 32) {
      Frag a;  ldfragA(a,  at + (rs + lm) * SS + k0 + 8 * g);
      Frag bf; ldfragB(bf, Vb + (ni * 16 + lm) * SS + k0 + 8 * g);
      acc.v = wmma_bf16(a, bf, acc.v);
    }
#pragma unroll
    for (int r = 0; r < 8; ++r) {
      long row = (long)b * LL + rs + r + 8 * g;
      int col = h * EE + ni * 16 + lm;
      Oc[row * (HH * EE) + col] = f2bf(acc.f[r]);
    }
  }
}

// ---------------------------------------------------------------------------
// LayerNorm over D=768 per row; writes f32 state + bf16 copy
__global__ __launch_bounds__(256)
void ln_k(const float* __restrict__ y, const float* __restrict__ gam,
          const float* __restrict__ bet, float* __restrict__ xo,
          u16* __restrict__ xob) {
  int row = blockIdx.x;
  int t = threadIdx.x;
  const float* yr = y + (long)row * DD;
  float v0 = yr[t], v1 = yr[t + 256], v2 = yr[t + 512];
  __shared__ float sred[256];
  sred[t] = v0 + v1 + v2;
  __syncthreads();
  for (int off = 128; off > 0; off >>= 1) {
    if (t < off) sred[t] += sred[t + off];
    __syncthreads();
  }
  float mean = sred[0] * (1.0f / DD);
  __syncthreads();
  float d0 = v0 - mean, d1 = v1 - mean, d2 = v2 - mean;
  sred[t] = d0 * d0 + d1 * d1 + d2 * d2;
  __syncthreads();
  for (int off = 128; off > 0; off >>= 1) {
    if (t < off) sred[t] += sred[t + off];
    __syncthreads();
  }
  float var = sred[0] * (1.0f / DD);
  float rs = rsqrtf(var + 1e-6f);
  float o0 = d0 * rs * gam[t]       + bet[t];
  float o1 = d1 * rs * gam[t + 256] + bet[t + 256];
  float o2 = d2 * rs * gam[t + 512] + bet[t + 512];
  float* xr = xo + (long)row * DD;
  u16*   xbr = xob + (long)row * DD;
  xr[t] = o0;        xbr[t] = f2bf(o0);
  xr[t + 256] = o1;  xbr[t + 256] = f2bf(o1);
  xr[t + 512] = o2;  xbr[t + 512] = f2bf(o2);
}

// ---------------------------------------------------------------------------
extern "C" void kernel_launch(void* const* d_in, const int* in_sizes, int n_in,
                              void* d_out, int out_size, void* d_ws, size_t ws_size,
                              hipStream_t stream) {
  const float* enc  = (const float*)d_in[0];
  const float* pos  = (const float*)d_in[1];
  const float* wq   = (const float*)d_in[2];
  const float* wk   = (const float*)d_in[3];
  const float* wv   = (const float*)d_in[4];
  const float* wo   = (const float*)d_in[5];
  const float* ln1g = (const float*)d_in[6];
  const float* ln1b = (const float*)d_in[7];
  const float* ln2g = (const float*)d_in[8];
  const float* ln2b = (const float*)d_in[9];
  const float* w1   = (const float*)d_in[10];
  const float* w2   = (const float*)d_in[11];

  char* ws = (char*)d_ws;
  size_t off = 0;
  auto alloc = [&](size_t bytes) -> void* {
    void* p = ws + off;
    off += (bytes + 255) & ~(size_t)255;
    return p;
  };

  u16* enc_b = (u16*)alloc((size_t)M2 * DD * 2);
  u16* wqT   = (u16*)alloc((size_t)NBLK * HH * EE * DD * 2);
  u16* wkT   = (u16*)alloc((size_t)NBLK * HH * EE * DD * 2);
  u16* wvT   = (u16*)alloc((size_t)NBLK * HH * EE * DD * 2);
  u16* woT   = (u16*)alloc((size_t)NBLK * DD * (HH * EE) * 2);
  u16* w1T   = (u16*)alloc((size_t)NBLK * DFF * DD * 2);
  u16* w2T   = (u16*)alloc((size_t)NBLK * DD * DFF * 2);
  float* xf  = (float*)alloc((size_t)M1 * DD * 4);
  u16* xb    = (u16*)alloc((size_t)M1 * DD * 2);
  u16* Qb    = (u16*)alloc((size_t)HH * M1 * EE * 2);
  u16* Kb    = (u16*)alloc((size_t)HH * M2 * EE * 2);
  u16* Vtb   = (u16*)alloc((size_t)HH * BB * EE * SS * 2);
  u16* Oc    = (u16*)alloc((size_t)M1 * (HH * EE) * 2);
  float* yb  = (float*)alloc((size_t)M1 * DD * 4);
  u16* hb    = (u16*)alloc((size_t)M1 * DFF * 2);

  // preprocessing
  {
    long n = (long)M2 * DD;
    cvt_bf16_k<<<(n + 255) / 256, 256, 0, stream>>>(enc, enc_b, n);
  }
  {
    long t = (long)NBLK * HH * DD * EE;
    tcvt_bf16_k<<<(t + 255) / 256, 256, 0, stream>>>(wq, wqT, DD, EE, t);
    tcvt_bf16_k<<<(t + 255) / 256, 256, 0, stream>>>(wk, wkT, DD, EE, t);
    tcvt_bf16_k<<<(t + 255) / 256, 256, 0, stream>>>(wv, wvT, DD, EE, t);
  }
  {
    long t = (long)NBLK * (HH * EE) * DD;
    tcvt_bf16_k<<<(t + 255) / 256, 256, 0, stream>>>(wo, woT, HH * EE, DD, t);
  }
  {
    long t = (long)NBLK * DD * DFF;
    tcvt_bf16_k<<<(t + 255) / 256, 256, 0, stream>>>(w1, w1T, DD, DFF, t);
    tcvt_bf16_k<<<(t + 255) / 256, 256, 0, stream>>>(w2, w2T, DFF, DD, t);
  }
  {
    long n = (long)M1 * DD;
    posinit_k<<<(n + 255) / 256, 256, 0, stream>>>(pos, xf, xb);
  }

  const size_t attn_smem = (size_t)LL * SS * 4 + (size_t)LL * SS * 2 + 256 * 4 + LL * 4;

  for (int i = 0; i < NBLK; ++i) {
    const u16* wqTi = wqT + (size_t)i * HH * EE * DD;
    const u16* wkTi = wkT + (size_t)i * HH * EE * DD;
    const u16* wvTi = wvT + (size_t)i * HH * EE * DD;
    const u16* woTi = woT + (size_t)i * DD * (HH * EE);
    const u16* w1Ti = w1T + (size_t)i * DFF * DD;
    const u16* w2Ti = w2T + (size_t)i * DD * DFF;

    // Q projection: [4096,768]x[768,32] per head -> Qb [H][4096][32] bf16
    gemm_k<1, 0, 2><<<dim3(M1 / 256, 1, HH), 128, 0, stream>>>(
        xb, wqTi, Qb, nullptr, M1, EE, DD,
        0L, (long)EE * DD, (long)M1 * EE);
    // K projection: [32768,768]x[768,32] per head -> Kb [H][32768][32] bf16
    gemm_k<1, 0, 2><<<dim3(M2 / 256, 1, HH), 128, 0, stream>>>(
        enc_b, wkTi, Kb, nullptr, M2, EE, DD,
        0L, (long)EE * DD, (long)M2 * EE);
    // V projection, stored transposed per (b,h): Vtb [H][B][32][512] bf16
    gemm_k<2, 0, 2><<<dim3(M2 / 256, 1, HH), 128, 0, stream>>>(
        enc_b, wvTi, Vtb, nullptr, M2, EE, DD,
        0L, (long)EE * DD, (long)BB * EE * SS);
    // fused attention -> Oc [4096][384] bf16
    attn_k<<<dim3(BB * HH), 256, attn_smem, stream>>>(Qb, Kb, Vtb, Oc);
    // output projection + residual: y = Oc @ wo + x  (wave tile 64x64)
    gemm_k<0, 1, 4><<<dim3(M1 / 256, DD / 64, 1), 128, 0, stream>>>(
        Oc, woTi, yb, xf, M1, DD, HH * EE, 0L, 0L, 0L);
    // LN1 -> x (f32 + bf16)
    ln_k<<<dim3(M1), 256, 0, stream>>>(yb, ln1g + (size_t)i * DD,
                                       ln1b + (size_t)i * DD, xf, xb);
    // FFN1 + gelu -> hb [4096][3072] bf16  (wave tile 64x64)
    gemm_k<1, 2, 4><<<dim3(M1 / 256, DFF / 64, 1), 128, 0, stream>>>(
        xb, w1Ti, hb, nullptr, M1, DFF, DD, 0L, 0L, 0L);
    // FFN2 + residual: y = hb @ w2 + x  (wave tile 64x64)
    gemm_k<0, 1, 4><<<dim3(M1 / 256, DD / 64, 1), 128, 0, stream>>>(
        hb, w2Ti, yb, xf, M1, DD, DFF, 0L, 0L, 0L);
    // LN2 -> x
    ln_k<<<dim3(M1), 256, 0, stream>>>(yb, ln2g + (size_t)i * DD,
                                       ln2b + (size_t)i * DD, xf, xb);
  }

  hipMemcpyAsync(d_out, xf, (size_t)M1 * DD * 4, hipMemcpyDeviceToDevice, stream);
}
